// TransformerLayer_64106681860296
// MI455X (gfx1250) — matrix operand order repaired and checked by reading, compiled, not run on hardware
//
#include <hip/hip_runtime.h>
#include <hip/hip_bf16.h>
#include <stdint.h>

// ---------------------------------------------------------------------------
// Transformer layer (B=2, S=2048, D=1024, H=16, dk=64, FF=4096) for gfx1250.
// bf16 WMMA everywhere; fp32 accumulation; online-softmax attention.
// CDNA5 data paths: TDM (tensor_load_to_lds) for strided weight/K tiles,
// GLOBAL_LOAD_ASYNC_TO_LDS for activation tiles, double-buffered LDS.
// ---------------------------------------------------------------------------

#define BATCH 2
#define SEQ   2048
#define DMODEL 1024
#define NHEAD 16
#define DK    64
#define DFF   4096
#define MROWS (BATCH * SEQ)          // 4096

typedef __attribute__((ext_vector_type(16))) __bf16 bfrag;   // WMMA A/B operand
typedef __attribute__((ext_vector_type(8)))  __bf16 bhalf8;  // 16-byte bf16 chunk
typedef __attribute__((ext_vector_type(8)))  float  v8f;     // WMMA C/D operand
typedef __attribute__((ext_vector_type(4)))  unsigned tdm_g0;  // D# group 0
typedef __attribute__((ext_vector_type(8)))  int      tdm_g1;  // D# group 1

__device__ __forceinline__ bfrag make_frag(const __bf16* p0, const __bf16* p1) {
  bhalf8 lo = *reinterpret_cast<const bhalf8*>(p0);
  bhalf8 hi = *reinterpret_cast<const bhalf8*>(p1);
  return __builtin_shufflevector(lo, hi, 0,1,2,3,4,5,6,7,8,9,10,11,12,13,14,15);
}

__device__ __forceinline__ v8f wmma_bf16(bfrag a, bfrag b, v8f c) {
  // D = A(16x32 bf16) * B(32x16 bf16) + C(16x16 f32)
  return __builtin_amdgcn_wmma_f32_16x16x32_bf16(
      /*neg_a=*/false, a, /*neg_b=*/false, b,
      /*c_mod=*/(short)0, c, /*reuse_a=*/false, /*reuse_b=*/false);
}

// xor-butterfly within 16-lane halves via DS_SWIZZLE (immediate pattern:
// offset[15]=0 group-of-32, xor_mask=M, or=0, and=0x1F).  For M<16 this is
// exactly __shfl_xor(x, M, 16) but with no per-use VALU address setup.
template<int M>
__device__ __forceinline__ float swz_xor16(float x) {
  return __int_as_float(
      __builtin_amdgcn_ds_swizzle(__float_as_int(x), (M << 10) | 0x1f));
}
__device__ __forceinline__ float redmax16(float x) {
  x = fmaxf(x, swz_xor16<8>(x));
  x = fmaxf(x, swz_xor16<4>(x));
  x = fmaxf(x, swz_xor16<2>(x));
  x = fmaxf(x, swz_xor16<1>(x));
  return x;
}
__device__ __forceinline__ float redsum16(float x) {
  x += swz_xor16<8>(x);
  x += swz_xor16<4>(x);
  x += swz_xor16<2>(x);
  x += swz_xor16<1>(x);
  return x;
}

// CDNA5 async copy: global -> LDS, 16B per lane, tracked by ASYNCcnt.
// dst_lds is the LDS byte offset (kernels below use only dynamic LDS so the
// allocation starts at offset 0).
__device__ __forceinline__ void async_ld_b128(unsigned dst_lds, const void* src) {
  asm volatile("global_load_async_to_lds_b128 %0, %1, off"
               :
               : "v"(dst_lds), "v"((unsigned long long)(uintptr_t)src)
               : "memory");
}
__device__ __forceinline__ void wait_async0() {
  asm volatile("s_wait_asynccnt 0x0" ::: "memory");
}

// CDNA5 Tensor Data Mover: DMA a 2D bf16 tile (tile_d0 elems x tile_d1 rows,
// row stride = stride_elems) from global memory into LDS at lds_byte, with
// hardware LDS padding (pad after every 8*2^pad_interval bytes).  Issued once
// per wave (EXEC ignored!) and tracked by TENSORcnt -- callers must gate with
// a *scalar* leader branch so only one wave per block issues it.
__device__ __forceinline__ void tdm_load_2d(
    unsigned lds_byte, const void* gaddr,
    unsigned tile_d0, unsigned tile_d1,
    unsigned long long stride_elems,
    unsigned pad_interval, unsigned pad_amount) {
  unsigned long long ga = (unsigned long long)(uintptr_t)gaddr;
  tdm_g0 g0;
  g0[0] = 1u;                                             // count=1, user D#
  g0[1] = lds_byte;                                       // lds_addr
  g0[2] = (unsigned)(ga & 0xFFFFFFFFu);                   // global_addr[31:0]
  g0[3] = (unsigned)((ga >> 32) & 0x01FFFFFFu) | (2u << 30);  // [56:32] | type=2
  tdm_g1 g1;
  g1[0] = (int)((1u << 16) |                              // data_size = 2 bytes
                (1u << 20) |                              // pad_enable
                (pad_interval << 22) | (pad_amount << 25));
  g1[1] = (int)((tile_d0 & 0xFFFFu) << 16);               // tensor_dim0 lo16
  g1[2] = (int)(((tile_d0 >> 16) & 0xFFFFu) |             // tensor_dim0 hi16
                ((tile_d1 & 0xFFFFu) << 16));             // tensor_dim1 lo16
  g1[3] = (int)(((tile_d1 >> 16) & 0xFFFFu) |             // tensor_dim1 hi16
                ((tile_d0 & 0xFFFFu) << 16));             // tile_dim0
  g1[4] = (int)(tile_d1 & 0xFFFFu);                       // tile_dim1 (dim2=0)
  g1[5] = (int)(unsigned)(stride_elems & 0xFFFFFFFFu);    // dim0_stride lo32
  g1[6] = (int)(unsigned)((stride_elems >> 32) & 0xFFFFu);// dim0_stride hi16
  g1[7] = 0;                                              // dim1_stride unused
  asm volatile("tensor_load_to_lds %0, %1" : : "s"(g0), "s"(g1) : "memory");
}
__device__ __forceinline__ void wait_tensor0() {
  __builtin_amdgcn_s_wait_tensorcnt(0);
}

// ---------------------------------------------------------------------------
// fp32 -> bf16 conversion kernels
// ---------------------------------------------------------------------------
__global__ __launch_bounds__(256) void cvt_bf16_kernel(
    const float* __restrict__ in, __bf16* __restrict__ out, int n) {
  int i = blockIdx.x * 256 + threadIdx.x;
  if (i < n) out[i] = (__bf16)in[i];
}

// in: [K][N] fp32 row-major  ->  out: [N][K] bf16 row-major (i.e. B^T)
__global__ __launch_bounds__(256) void cvt_bf16_t_kernel(
    const float* __restrict__ in, __bf16* __restrict__ out, int K, int N) {
  int i = blockIdx.x * 256 + threadIdx.x;
  if (i < K * N) {
    int nn = i / K;
    int kk = i - nn * K;
    out[i] = (__bf16)in[(size_t)kk * N + nn];
  }
}

// ---------------------------------------------------------------------------
// Generic bf16 GEMM:  out[M][N] = A[M][K] @ B[K][N] (+bias) (relu) (+resid)
// B supplied transposed: Bt[N][K].  BM=128, BN=128, BK=32; 8 waves/block,
// each wave owns a 32x64 tile (2x4 WMMA tiles).
// A tiles: per-lane async-DMA.  Bt tiles: one TDM descriptor per K-step
// (issued by the scalar-branch leader wave only).  Double buffered.
// ---------------------------------------------------------------------------
#define GBM 128
#define GBN 128
#define GBK 32
#define LDS_STRIDE 40                 // 32 bf16 data + 8 pad
#define GA_BYTES (GBM * LDS_STRIDE * 2)      // 10240 per A buffer
#define GB_BYTES (GBN * LDS_STRIDE * 2)      // 10240 per B buffer
#define GB_BASE  (2 * GA_BYTES)              // B region after 2 A buffers
#define GEMM_LDS (2 * GA_BYTES + 2 * GB_BYTES)  // 40960 bytes

template<bool RELU, bool RESID, bool OUT_BF16>
__global__ __launch_bounds__(256) void gemm_bf16_kernel(
    const __bf16* __restrict__ A, const __bf16* __restrict__ Bt,
    const float* __restrict__ bias, const float* __restrict__ resid,
    float* __restrict__ outF, __bf16* __restrict__ outB,
    int M, int N, int K) {
  extern __shared__ __align__(16) char smem_raw[];
  __bf16* ldsA = (__bf16*)smem_raw;               // [2][GBM*LDS_STRIDE]
  __bf16* ldsB = (__bf16*)(smem_raw + GB_BASE);   // [2][GBN*LDS_STRIDE]

  const int tid  = threadIdx.x;
  const int lane = tid & 31;
  const int w    = tid >> 5;      // wave 0..7
  const int wy   = w & 3;         // row group (4 x 32 rows)
  const int wx   = w >> 2;        // col group (2 x 64 cols)
  const int r    = lane & 15;
  const int hi   = lane >> 4;     // half-wave select
  const int blockN = blockIdx.x * GBN;
  const int blockM = blockIdx.y * GBM;
  // Scalar (SGPR) leader predicate => s_cbranch, so only wave 0 issues TDM.
  const bool leader = (__builtin_amdgcn_readfirstlane(w) == 0);

  // Stage one K-tile (32 cols) of A and Bt into LDS buffer `buf`.
  auto stage = [&](int kt, int buf) {
    // A: 512 x 16B chunks via per-lane async DMA (ASYNCcnt)
#pragma unroll
    for (int i = 0; i < 2; ++i) {
      int idx = tid + i * 256;
      int row = idx >> 2, q = idx & 3;
      unsigned off = (unsigned)(buf * GA_BYTES + (row * LDS_STRIDE + q * 8) * 2);
      async_ld_b128(off, A + (size_t)(blockM + row) * K + kt * GBK + q * 8);
    }
    // Bt: one TDM descriptor for the whole 32x128 tile (TENSORcnt), with
    // hardware pad: 16 DWORDs (one 64B row) -> +4 DWORDs => 40-elem stride.
    if (leader) {
      tdm_load_2d((unsigned)(GB_BASE + buf * GB_BYTES),
                  Bt + (size_t)blockN * K + (size_t)kt * GBK,
                  /*tile_d0=*/GBK, /*tile_d1=*/GBN,
                  /*stride_elems=*/(unsigned long long)K,
                  /*pad_interval=*/3, /*pad_amount=*/3);
    }
  };

  v8f acc[2][4];
#pragma unroll
  for (int i = 0; i < 2; ++i)
#pragma unroll
    for (int j = 0; j < 4; ++j) acc[i][j] = {};

  const int kIters = K / GBK;
  stage(0, 0);
  wait_async0();
  if (leader) wait_tensor0();
  __syncthreads();

  for (int kt = 0; kt < kIters; ++kt) {
    const int cur = kt & 1;
    if (kt + 1 < kIters) stage(kt + 1, cur ^ 1);   // overlap DMA with compute
    if (kt + 2 < kIters) {                         // speculative L2 prefetch
      int row = tid >> 1;
      __builtin_prefetch(A + (size_t)(blockM + row) * K + (kt + 2) * GBK, 0, 3);
      __builtin_prefetch(Bt + (size_t)(blockN + row) * K + (kt + 2) * GBK, 0, 3);
    }

    const __bf16* aBuf = ldsA + (size_t)cur * GBM * LDS_STRIDE;
    const __bf16* bBuf = ldsB + (size_t)cur * GBN * LDS_STRIDE;

    // A fragments: 16x32 (lanes 0-15: K 0-15; lanes 16-31: K 16-31)
    bfrag a[2], b[4];
#pragma unroll
    for (int i = 0; i < 2; ++i) {
      const __bf16* base = aBuf + (wy * 32 + i * 16 + r) * LDS_STRIDE;
      a[i] = make_frag(base + hi * 16, base + hi * 16 + 8);
    }
    // B fragments: 32x16, lane n=L&15, K chunk = hi*16 .. +16 (contiguous)
#pragma unroll
    for (int j = 0; j < 4; ++j) {
      const __bf16* base = bBuf + (wx * 64 + j * 16 + r) * LDS_STRIDE + hi * 16;
      b[j] = make_frag(base, base + 8);
    }
#pragma unroll
    for (int i = 0; i < 2; ++i)
#pragma unroll
      for (int j = 0; j < 4; ++j)
        acc[i][j] = wmma_bf16(a[i], b[j], acc[i][j]);

    wait_async0();               // next buffer's async loads done
    if (leader) wait_tensor0();  // next buffer's TDM done
    __syncthreads();             // all waves done reading `cur`
  }

  // Epilogue.  C layout: element e -> row m = e + hi*8, col n = L&15.
#pragma unroll
  for (int i = 0; i < 2; ++i)
#pragma unroll
    for (int j = 0; j < 4; ++j)
#pragma unroll
      for (int e = 0; e < 8; ++e) {
        int m = blockM + wy * 32 + i * 16 + e + hi * 8;
        int n = blockN + wx * 64 + j * 16 + r;
        float val = acc[i][j][e];
        val += bias[n];
        if (RELU)  val = fmaxf(val, 0.0f);
        if (RESID) val += resid[(size_t)m * N + n];
        if (OUT_BF16) outB[(size_t)m * N + n] = (__bf16)val;
        else          outF[(size_t)m * N + n] = val;
      }
}

// ---------------------------------------------------------------------------
// Flash attention.  Block = (b, h, group of 8 q-tiles); wave = one 16-query
// tile.  Streams keys in 32-wide chunks; scores = 2 WMMAs over dk=64,
// online softmax, PV = 4 WMMAs.  K tiles staged with TDM (strided rows +
// hardware pad), V tiles transposed through VGPRs; both double buffered so
// chunk c+1's DMA overlaps chunk c's compute (one barrier per chunk).
//   q indexed head-major (reference's un-transposed reshape):
//     q[b,h,s,:] = qproj_flat[((b*H+h)*S + s)*DK ..]
//   k,v indexed seq-major:  k[b,h,s,:] = kproj_flat[(b*S+s)*D + h*DK ..]
// ---------------------------------------------------------------------------
#define KS_STRIDE 72   // 64 dims + 8 pad  (TDM pad: 32 DW -> +4 DW)
#define VS_STRIDE 40   // 32 keys + 8 pad
#define PS_STRIDE 40   // 32 probs + 8 pad
#define ATT_KS_BYTES (32 * KS_STRIDE * 2)      // 4608 per buffer
#define ATT_VS_BYTES (64 * VS_STRIDE * 2)      // 5120 per buffer
#define ATT_PS_BYTES (8 * 16 * PS_STRIDE * 2)  // 10240
#define ATT_VS_BASE  (2 * ATT_KS_BYTES)                    // 9216
#define ATT_PS_BASE  (ATT_VS_BASE + 2 * ATT_VS_BYTES)      // 19456
#define ATT_LDS      (ATT_PS_BASE + ATT_PS_BYTES)          // 29696

__global__ __launch_bounds__(256) void attention_kernel(
    const __bf16* __restrict__ qm, const __bf16* __restrict__ km,
    const __bf16* __restrict__ vm, __bf16* __restrict__ om) {
  extern __shared__ __align__(16) char smem_raw[];
  __bf16* kS = (__bf16*)smem_raw;                       // [2][32][72]
  __bf16* vS = (__bf16*)(smem_raw + ATT_VS_BASE);       // [2][64][40]
  __bf16* pS = (__bf16*)(smem_raw + ATT_PS_BASE);       // [8][16][40]

  const int tid  = threadIdx.x;
  const int lane = tid & 31;
  const int w    = tid >> 5;
  const int r    = lane & 15;
  const int hi   = lane >> 4;
  const bool leader = (__builtin_amdgcn_readfirstlane(w) == 0);

  const int bx = blockIdx.x;          // 0..511
  const int qg = bx & 15;             // 16 q groups of 128 queries
  const int h  = (bx >> 4) & 15;
  const int b  = bx >> 8;
  const int q0 = (qg * 8 + w) * 16;   // first query row of this wave

  // Load q fragments once (A layout, two K-halves of dk=64)
  const size_t qbase = (((size_t)(b * NHEAD + h)) * SEQ + q0) * DK;
  bfrag qf[2];
#pragma unroll
  for (int ks = 0; ks < 2; ++ks) {
    const __bf16* base = qm + qbase + (size_t)r * DK + ks * 32;
    qf[ks] = make_frag(base + hi * 16, base + hi * 16 + 8);
  }

  // Stage one 32-key chunk: K tile via TDM (leader wave), V tile transposed
  // through VGPRs into vS[buf].
  auto stage = [&](int c, int buf) {
    const int key0 = c * 32;
    if (leader) {
      tdm_load_2d((unsigned)(buf * ATT_KS_BYTES),
                  km + ((size_t)(b * SEQ + key0)) * DMODEL + h * DK,
                  /*tile_d0=*/DK, /*tile_d1=*/32,
                  /*stride_elems=*/(unsigned long long)DMODEL,
                  /*pad_interval=*/4, /*pad_amount=*/3);
    }
    int kk  = tid >> 3;              // key 0..31
    int col = (tid & 7) * 8;         // dim chunk
    const size_t gb = ((size_t)(b * SEQ + key0 + kk)) * DMODEL + h * DK + col;
    bhalf8 dv8 = *reinterpret_cast<const bhalf8*>(vm + gb);
    __bf16* vb = vS + buf * (64 * VS_STRIDE);
#pragma unroll
    for (int e = 0; e < 8; ++e)
      vb[(col + e) * VS_STRIDE + kk] = dv8[e];
  };

  v8f oacc[4];
  oacc[0] = {}; oacc[1] = {}; oacc[2] = {}; oacc[3] = {};
  float mrow[8], lrow[8];
#pragma unroll
  for (int e = 0; e < 8; ++e) { mrow[e] = -1e30f; lrow[e] = 0.0f; }
  const float scale = 0.125f;  // 1/sqrt(64)

  const int NC = SEQ / 32;     // 64 chunks
  stage(0, 0);
  if (leader) wait_tensor0();
  __syncthreads();

  for (int c = 0; c < NC; ++c) {
    const int buf = c & 1;
    if (c + 1 < NC) stage(c + 1, buf ^ 1);   // overlap next chunk's staging

    const __bf16* kB = kS + buf * (32 * KS_STRIDE);
    const __bf16* vB = vS + buf * (64 * VS_STRIDE);

    // Scores: two 16-key tiles, each = 2 WMMAs over dk
    float p[2][8];
#pragma unroll
    for (int nt = 0; nt < 2; ++nt) {
      const __bf16* kb = kB + (nt * 16 + r) * KS_STRIDE;
      bfrag kf0 = make_frag(kb + hi * 16, kb + hi * 16 + 8);            // dims 0-31
      bfrag kf1 = make_frag(kb + 32 + hi * 16, kb + 32 + hi * 16 + 8);  // dims 32-63
      v8f s = {};
      s = wmma_bf16(qf[0], kf0, s);
      s = wmma_bf16(qf[1], kf1, s);
#pragma unroll
      for (int e = 0; e < 8; ++e) p[nt][e] = s[e] * scale;
    }

    // Online softmax (row = e + hi*8 lives in this 16-lane half)
#pragma unroll
    for (int e = 0; e < 8; ++e) {
      float nm = fmaxf(mrow[e], redmax16(fmaxf(p[0][e], p[1][e])));
      float al = __expf(mrow[e] - nm);
      mrow[e] = nm;
      float e0 = __expf(p[0][e] - nm);
      float e1 = __expf(p[1][e] - nm);
      p[0][e] = e0; p[1][e] = e1;
      lrow[e] = lrow[e] * al + redsum16(e0 + e1);
      oacc[0][e] *= al; oacc[1][e] *= al; oacc[2][e] *= al; oacc[3][e] *= al;
    }

    // Re-layout P (C layout) -> A fragment via per-wave LDS (same-wave DS
    // ops are in-order, no barrier needed)
    __bf16* pw = pS + w * (16 * PS_STRIDE);
#pragma unroll
    for (int e = 0; e < 8; ++e) {
      int mloc = e + hi * 8;
      __bf16* pp = pw + mloc * PS_STRIDE + r;
      pp[0]  = (__bf16)p[0][e];
      pp[16] = (__bf16)p[1][e];
    }
    const __bf16* pb = pw + r * PS_STRIDE;
    bfrag pf = make_frag(pb + hi * 16, pb + hi * 16 + 8);

    // PV: N = dk tiled into 4 x 16
#pragma unroll
    for (int nt = 0; nt < 4; ++nt) {
      const __bf16* vb = vB + (nt * 16 + r) * VS_STRIDE + hi * 16;
      bfrag vf = make_frag(vb, vb + 8);
      oacc[nt] = wmma_bf16(pf, vf, oacc[nt]);
    }

    if (leader) wait_tensor0();  // next chunk's TDM complete
    __syncthreads();             // next chunk's vS stores visible; buf free
  }

  // Normalize and scatter to [B, S, H*DK] (reference's output transpose)
#pragma unroll
  for (int e = 0; e < 8; ++e) {
    float inv = 1.0f / lrow[e];
    int m = e + hi * 8;
    size_t ob = ((size_t)(b * SEQ + q0 + m)) * DMODEL + h * DK + r;
#pragma unroll
    for (int nt = 0; nt < 4; ++nt)
      om[ob + nt * 16] = (__bf16)(oacc[nt][e] * inv);
  }
}

// ---------------------------------------------------------------------------
// LayerNorm over rows of D=1024; optional bf16 shadow copy for next GEMM.
// ---------------------------------------------------------------------------
__global__ __launch_bounds__(256) void layernorm_kernel(
    const float* __restrict__ x, const float* __restrict__ sc,
    const float* __restrict__ bi, float* __restrict__ outF,
    __bf16* __restrict__ outB, int D) {
  __shared__ float red[2][8];
  const int row = blockIdx.x;
  const float* xr = x + (size_t)row * D;
  float s = 0.0f, s2 = 0.0f;
  for (int i = threadIdx.x; i < D; i += 256) {
    float v = xr[i]; s += v; s2 += v * v;
  }
#pragma unroll
  for (int m = 16; m; m >>= 1) {
    s  += __shfl_xor(s, m, 32);
    s2 += __shfl_xor(s2, m, 32);
  }
  if ((threadIdx.x & 31) == 0) {
    red[0][threadIdx.x >> 5] = s;
    red[1][threadIdx.x >> 5] = s2;
  }
  __syncthreads();
  float ts = 0.0f, ts2 = 0.0f;
#pragma unroll
  for (int i = 0; i < 8; ++i) { ts += red[0][i]; ts2 += red[1][i]; }
  float mean = ts / (float)D;
  float var  = ts2 / (float)D - mean * mean;
  float inv  = rsqrtf(var + 1e-5f);
  for (int i = threadIdx.x; i < D; i += 256) {
    float v = (xr[i] - mean) * inv * sc[i] + bi[i];
    if (outF) outF[(size_t)row * D + i] = v;
    if (outB) outB[(size_t)row * D + i] = (__bf16)v;
  }
}

// ---------------------------------------------------------------------------
// Host-side orchestration
// ---------------------------------------------------------------------------
extern "C" void kernel_launch(void* const* d_in, const int* in_sizes, int n_in,
                              void* d_out, int out_size, void* d_ws, size_t ws_size,
                              hipStream_t stream) {
  const float* x    = (const float*)d_in[0];
  const float* Wq   = (const float*)d_in[1];
  const float* bq   = (const float*)d_in[2];
  const float* Wk   = (const float*)d_in[3];
  const float* bk   = (const float*)d_in[4];
  const float* Wv   = (const float*)d_in[5];
  const float* bv   = (const float*)d_in[6];
  const float* Wo   = (const float*)d_in[7];
  const float* bo   = (const float*)d_in[8];
  const float* ln1s = (const float*)d_in[9];
  const float* ln1b = (const float*)d_in[10];
  const float* ln2s = (const float*)d_in[11];
  const float* ln2b = (const float*)d_in[12];
  const float* W1   = (const float*)d_in[13];
  const float* b1   = (const float*)d_in[14];
  const float* W2   = (const float*)d_in[15];
  const float* b2   = (const float*)d_in[16];
  float* out = (float*)d_out;

  // Workspace carve-up (MB-aligned offsets, ~152 MB total)
  char* ws = (char*)d_ws;
  const size_t MB = 1024 * 1024;
  __bf16* x_bf    = (__bf16*)(ws + 0 * MB);     // 8 MB
  __bf16* WqT     = (__bf16*)(ws + 8 * MB);     // 2 MB
  __bf16* WkT     = (__bf16*)(ws + 10 * MB);    // 2 MB
  __bf16* WvT     = (__bf16*)(ws + 12 * MB);    // 2 MB
  __bf16* WoT     = (__bf16*)(ws + 14 * MB);    // 2 MB
  __bf16* W1T     = (__bf16*)(ws + 16 * MB);    // 8 MB
  __bf16* W2T     = (__bf16*)(ws + 24 * MB);    // 8 MB
  __bf16* q_bf    = (__bf16*)(ws + 32 * MB);    // 8 MB
  __bf16* k_bf    = (__bf16*)(ws + 40 * MB);    // 8 MB
  __bf16* v_bf    = (__bf16*)(ws + 48 * MB);    // 8 MB
  __bf16* attn_bf = (__bf16*)(ws + 56 * MB);    // 8 MB
  float*  y1      = (float* )(ws + 64 * MB);    // 16 MB
  float*  xn_f    = (float* )(ws + 80 * MB);    // 16 MB
  __bf16* xn_bf   = (__bf16*)(ws + 96 * MB);    // 8 MB
  __bf16* h_bf    = (__bf16*)(ws + 104 * MB);   // 32 MB
  float*  y2      = (float* )(ws + 136 * MB);   // 16 MB
  (void)in_sizes; (void)n_in; (void)out_size; (void)ws_size;

  const int NX = MROWS * DMODEL;  // 4M elements

  // 1) Precision conversion
  cvt_bf16_kernel<<<NX / 256, 256, 0, stream>>>(x, x_bf, NX);
  cvt_bf16_t_kernel<<<(DMODEL * DMODEL) / 256, 256, 0, stream>>>(Wq, WqT, DMODEL, DMODEL);
  cvt_bf16_t_kernel<<<(DMODEL * DMODEL) / 256, 256, 0, stream>>>(Wk, WkT, DMODEL, DMODEL);
  cvt_bf16_t_kernel<<<(DMODEL * DMODEL) / 256, 256, 0, stream>>>(Wv, WvT, DMODEL, DMODEL);
  cvt_bf16_t_kernel<<<(DMODEL * DMODEL) / 256, 256, 0, stream>>>(Wo, WoT, DMODEL, DMODEL);
  cvt_bf16_t_kernel<<<(DMODEL * DFF) / 256, 256, 0, stream>>>(W1, W1T, DMODEL, DFF);
  cvt_bf16_t_kernel<<<(DFF * DMODEL) / 256, 256, 0, stream>>>(W2, W2T, DFF, DMODEL);

  // 2) QKV projections (bf16 out)
  dim3 gemmD(DMODEL / GBN, MROWS / GBM);   // (8, 32)
  gemm_bf16_kernel<false, false, true><<<gemmD, 256, GEMM_LDS, stream>>>(
      x_bf, WqT, bq, nullptr, nullptr, q_bf, MROWS, DMODEL, DMODEL);
  gemm_bf16_kernel<false, false, true><<<gemmD, 256, GEMM_LDS, stream>>>(
      x_bf, WkT, bk, nullptr, nullptr, k_bf, MROWS, DMODEL, DMODEL);
  gemm_bf16_kernel<false, false, true><<<gemmD, 256, GEMM_LDS, stream>>>(
      x_bf, WvT, bv, nullptr, nullptr, v_bf, MROWS, DMODEL, DMODEL);

  // 3) Attention: B*H*(S/128) blocks
  attention_kernel<<<BATCH * NHEAD * (SEQ / 128), 256, ATT_LDS, stream>>>(
      q_bf, k_bf, v_bf, attn_bf);

  // 4) Output projection + residual (f32 out), then LN1
  gemm_bf16_kernel<false, true, false><<<gemmD, 256, GEMM_LDS, stream>>>(
      attn_bf, WoT, bo, x, y1, nullptr, MROWS, DMODEL, DMODEL);
  layernorm_kernel<<<MROWS, 256, 0, stream>>>(y1, ln1s, ln1b, xn_f, xn_bf, DMODEL);

  // 5) FFN
  dim3 gemmF1(DFF / GBN, MROWS / GBM);     // (32, 32)
  gemm_bf16_kernel<true, false, true><<<gemmF1, 256, GEMM_LDS, stream>>>(
      xn_bf, W1T, b1, nullptr, nullptr, h_bf, MROWS, DFF, DMODEL);
  gemm_bf16_kernel<false, true, false><<<gemmD, 256, GEMM_LDS, stream>>>(
      h_bf, W2T, b2, xn_f, y2, nullptr, MROWS, DMODEL, DFF);

  // 6) LN2 -> final output (f32)
  layernorm_kernel<<<MROWS, 256, 0, stream>>>(y2, ln2s, ln2b, out, nullptr, DMODEL);
}
